// ImageEncoder_61512521614020
// MI455X (gfx1250) — matrix operand order repaired and checked by reading, not compiled, not run on hardware
//
#include <hip/hip_runtime.h>
#include <hip/hip_bf16.h>

#define B_ 8
#define S_ 1024
#define P_ 768
#define D_ 1024
#define H_ 16
#define L_ 8
#define HD_ 64
#define FF_ 4096

typedef __attribute__((ext_vector_type(16))) __bf16 v16bf;
typedef __attribute__((ext_vector_type(8)))  float  v8f;
typedef unsigned int   uint32;
typedef unsigned short u16;

union ABfrag { uint32 u[8]; v16bf v; };

__device__ __forceinline__ u16 f2bf(float f) {
  union { float f; uint32 u; } c; c.f = f;
  return (u16)((c.u + 0x7FFFu + ((c.u >> 16) & 1u)) >> 16);
}

__device__ __forceinline__ v8f zero8() {
  v8f z = {0.f,0.f,0.f,0.f,0.f,0.f,0.f,0.f};
  return z;
}

// ---------------------------------------------------------------------------
// Generic GEMM: C[M,N] = act(A[M,K] @ W[K,N] + bias)
// 128x128 block tile, BK=32, 8 waves in 4x2 grid, each wave 32x64 (2x4 WMMA).
// Ping-pong double-buffered LDS (one barrier per k-step) with register-staged
// global prefetch. All global indexing is 32-bit offsets from uniform bases
// (buffers < 2 GB) so loads lower to saddr+voffset form.
// A in LDS as [m][k] (bf16), W in LDS transposed as [n][k] (bf16) so both
// fragment types load as contiguous dwords per the CDNA5 WMMA VGPR layouts.
// flags bit0 = ReLU. outF (fp32) / outB (bf16) written if non-null.
// ---------------------------------------------------------------------------
template <bool A_BF16>
__global__ __launch_bounds__(256) void gemm_kernel(
    const void* __restrict__ A_, const float* __restrict__ W,
    const float* __restrict__ bias, float* __restrict__ outF,
    u16* __restrict__ outB, int M, int N, int K, int flags) {
  __shared__ u16 As[2][128][40];
  __shared__ u16 Bs[2][128][40];
  const int tid  = threadIdx.x;
  const int lane = tid & 31;
  const int wid  = tid >> 5;
  const int wm = wid >> 1, wn = wid & 1;
  const int ml = lane & 15, hi = lane >> 4;
  const int n0 = blockIdx.x * 128, m0 = blockIdx.y * 128;

  // loader thread mapping (constant across k-steps)
  const int arow = tid >> 3;        // A: rows arow + 32*i, cols ac4..ac4+3
  const int ac4  = (tid & 7) * 4;
  const int wrow = tid >> 5;        // W: rows wrow + 8*i, cols wc4..wc4+3
  const int wc4  = (tid & 31) * 4;

  // register staging for global->LDS prefetch
  uint2  apf_h[4];
  float4 apf_f[4];
  float4 wpf[4];

  auto load_tiles = [&](int k0) {
    if constexpr (A_BF16) {
      const u16* Ap = (const u16*)A_ + (m0 * K + k0 + ac4);
#pragma unroll
      for (int i = 0; i < 4; ++i)
        apf_h[i] = *(const uint2*)(Ap + (arow + 32 * i) * K);
    } else {
      const float* Ap = (const float*)A_ + (m0 * K + k0 + ac4);
#pragma unroll
      for (int i = 0; i < 4; ++i)
        apf_f[i] = *(const float4*)(Ap + (arow + 32 * i) * K);
    }
    const float* Wp = W + (k0 * N + n0 + wc4);
#pragma unroll
    for (int i = 0; i < 4; ++i)
      wpf[i] = *(const float4*)(Wp + (wrow + 8 * i) * N);
  };

  auto store_tiles = [&](int buf) {
#pragma unroll
    for (int i = 0; i < 4; ++i) {
      u16 h0, h1, h2, h3;
      if constexpr (A_BF16) {
        h0 = (u16)apf_h[i].x; h1 = (u16)(apf_h[i].x >> 16);
        h2 = (u16)apf_h[i].y; h3 = (u16)(apf_h[i].y >> 16);
      } else {
        h0 = f2bf(apf_f[i].x); h1 = f2bf(apf_f[i].y);
        h2 = f2bf(apf_f[i].z); h3 = f2bf(apf_f[i].w);
      }
      uint32* dst = (uint32*)&As[buf][arow + 32 * i][ac4];
      dst[0] = (uint32)h0 | ((uint32)h1 << 16);
      dst[1] = (uint32)h2 | ((uint32)h3 << 16);
    }
#pragma unroll
    for (int i = 0; i < 4; ++i) {
      int kr = wrow + 8 * i;
      Bs[buf][wc4 + 0][kr] = f2bf(wpf[i].x);
      Bs[buf][wc4 + 1][kr] = f2bf(wpf[i].y);
      Bs[buf][wc4 + 2][kr] = f2bf(wpf[i].z);
      Bs[buf][wc4 + 3][kr] = f2bf(wpf[i].w);
    }
  };

  v8f acc[2][4];
#pragma unroll
  for (int a = 0; a < 2; ++a)
#pragma unroll
    for (int b = 0; b < 4; ++b) acc[a][b] = zero8();

  const int nk = K >> 5;
  load_tiles(0);
  store_tiles(0);
  if (nk > 1) load_tiles(32);
  __syncthreads();

  for (int kt = 0; kt < nk; ++kt) {
    const int cur = kt & 1;
    ABfrag af[2], bf[4];
#pragma unroll
    for (int tm = 0; tm < 2; ++tm) {
      int am = wm * 32 + tm * 16 + ml;
      int ak = hi * 8;
#pragma unroll
      for (int j = 0; j < 4; ++j) {
        af[tm].u[j]     = *(const uint32*)&As[cur][am][ak + 2 * j];
        af[tm].u[4 + j] = *(const uint32*)&As[cur][am][16 + ak + 2 * j];
      }
    }
#pragma unroll
    for (int tn = 0; tn < 4; ++tn) {
      int bn = wn * 64 + tn * 16 + ml;
      int bk = hi * 8;
#pragma unroll
      for (int j = 0; j < 8; ++j)
        bf[tn].u[j] = *(const uint32*)&Bs[cur][bn][bk + 2 * j + (j >= 4 ? 8 : 0)];
    }
#pragma unroll
    for (int tm = 0; tm < 2; ++tm)
#pragma unroll
      for (int tn = 0; tn < 4; ++tn)
        acc[tm][tn] = __builtin_amdgcn_wmma_f32_16x16x32_bf16(
            false, af[tm].v, false, bf[tn].v, (short)0, acc[tm][tn], false, false);

    if (kt + 1 < nk) {
      store_tiles(1 - cur);                       // stage tile kt+1
      if (kt + 2 < nk) load_tiles((kt + 2) << 5); // prefetch tile kt+2
      __syncthreads();
    }
  }

  // ---- epilogue (32-bit offsets) ----
#pragma unroll
  for (int tm = 0; tm < 2; ++tm) {
#pragma unroll
    for (int tn = 0; tn < 4; ++tn) {
      int m = m0 + wm * 32 + tm * 16 + hi * 8;
      int n = n0 + wn * 64 + tn * 16 + ml;
      float bv = bias ? bias[n] : 0.f;
#pragma unroll
      for (int r = 0; r < 8; ++r) {
        float v = acc[tm][tn][r] + bv;
        if (flags & 1) v = fmaxf(v, 0.f);
        int o = (m + r) * N + n;
        if (outF) outF[o] = v;
        if (outB) outB[o] = f2bf(v);
      }
    }
  }
}

// ---------------------------------------------------------------------------
// RoPE (applied once post-embedding). Writes fp32 residual in place + bf16 copy.
// ---------------------------------------------------------------------------
__global__ __launch_bounds__(256) void rope_kernel(
    float* __restrict__ x, u16* __restrict__ xbf, const int* __restrict__ lengths) {
  int bs = blockIdx.x;
  int b = bs >> 10;           // S = 1024
  int s = bs & 1023;
  int pos = s - (lengths[b] >> 1);
  float fpos = (float)pos;
  int base = bs * D_;
  for (int i = threadIdx.x; i < D_ / 2; i += 256) {
    int p = i & 31;                       // pair index within head (HD/2 = 32)
    int d0 = ((i >> 5) << 6) + 2 * p;     // head*64 + 2p
    float freq = __expf(-(float)p * 0.28782313662425573f);  // ln(1e4)/32
    float ang = fpos * freq;
    float cs = __cosf(ang), sn = __sinf(ang);
    float x1 = x[base + d0], x2 = x[base + d0 + 1];
    float o1 = x1 * cs - x2 * sn;
    float o2 = x1 * sn + x2 * cs;
    x[base + d0] = o1; x[base + d0 + 1] = o2;
    xbf[base + d0] = f2bf(o1); xbf[base + d0 + 1] = f2bf(o2);
  }
}

// ---------------------------------------------------------------------------
// Flash attention: one block per (b, h, 64-row q block). bf16 Q/K/V in
// [B*S, D] layout (head h at cols h*64..h*64+63). Online softmax with key
// padding mask; O accumulated in fp32 WMMA accumulators; bf16 output.
// ---------------------------------------------------------------------------
__global__ __launch_bounds__(256) void flash_kernel(
    const u16* __restrict__ Q, const u16* __restrict__ Kb,
    const u16* __restrict__ Vb, u16* __restrict__ O,
    const int* __restrict__ lengths) {
  __shared__ u16 Qs[64][72];
  __shared__ u16 Ks[64][72];
  __shared__ u16 Vs[64][72];    // stored transposed: Vs[d][kv]
  __shared__ float Ssm[64][68];
  __shared__ u16 Ps[64][72];
  __shared__ float rowM[64], rowL[64], rowA[64];

  const int tid = threadIdx.x, lane = tid & 31, wid = tid >> 5;
  const int wm = wid >> 1, wn = wid & 1;       // 4x2 wave grid: 16q x 32 cols
  const int ml = lane & 15, hi = lane >> 4;
  const int q0 = blockIdx.x * 64;
  const int h  = blockIdx.y;
  const int b  = blockIdx.z;
  const int len = lengths[b];
  const int baseRow = b * S_;
  const int dcol0 = h * HD_;

  // load Q tile (64 x 64 halves); 32-bit offsets (buffers are 16 MB)
#pragma unroll
  for (int i = 0; i < 8; ++i) {
    int idx = tid + i * 256;
    int row = idx >> 5, c = (idx & 31) * 2;
    *(uint32*)&Qs[row][c] =
        *(const uint32*)(Q + ((baseRow + q0 + row) * D_ + dcol0 + c));
  }
  if (tid < 64) { rowM[tid] = -1e30f; rowL[tid] = 0.f; }

  v8f o0 = zero8(), o1 = zero8();
  __syncthreads();

  for (int kv0 = 0; kv0 < S_; kv0 += 64) {
    // load K (row-major) and V (transposed) tiles
#pragma unroll
    for (int i = 0; i < 8; ++i) {
      int idx = tid + i * 256;
      int row = idx >> 5, c = (idx & 31) * 2;
      uint32 kw_ = *(const uint32*)(Kb + ((baseRow + kv0 + row) * D_ + dcol0 + c));
      *(uint32*)&Ks[row][c] = kw_;
      uint32 vw_ = *(const uint32*)(Vb + ((baseRow + kv0 + row) * D_ + dcol0 + c));
      Vs[c][row]     = (u16)vw_;
      Vs[c + 1][row] = (u16)(vw_ >> 16);
    }
    __syncthreads();

    // S = Q @ K^T   (kdim = 64 -> 2 WMMA k-steps)
    v8f s0 = zero8(), s1 = zero8();
#pragma unroll
    for (int kk = 0; kk < 64; kk += 32) {
      ABfrag af, b0, b1;
      int am = wm * 16 + ml;
      int ak = kk + hi * 8;
#pragma unroll
      for (int j = 0; j < 4; ++j) {
        af.u[j]     = *(const uint32*)&Qs[am][ak + 2 * j];
        af.u[4 + j] = *(const uint32*)&Qs[am][ak + 16 + 2 * j];
      }
      int bk = kk + hi * 8;
      int bn0 = wn * 32 + ml, bn1 = bn0 + 16;
#pragma unroll
      for (int j = 0; j < 8; ++j) {
        b0.u[j] = *(const uint32*)&Ks[bn0][bk + 2 * j + (j >= 4 ? 8 : 0)];
        b1.u[j] = *(const uint32*)&Ks[bn1][bk + 2 * j + (j >= 4 ? 8 : 0)];
      }
      s0 = __builtin_amdgcn_wmma_f32_16x16x32_bf16(false, af.v, false, b0.v, (short)0, s0, false, false);
      s1 = __builtin_amdgcn_wmma_f32_16x16x32_bf16(false, af.v, false, b1.v, (short)0, s1, false, false);
    }
    // store scaled scores (1/sqrt(64) = 0.125)
    {
      int sr = wm * 16 + hi * 8;
      int sc0 = wn * 32 + ml, sc1 = sc0 + 16;
#pragma unroll
      for (int r = 0; r < 8; ++r) {
        Ssm[sr + r][sc0] = s0[r] * 0.125f;
        Ssm[sr + r][sc1] = s1[r] * 0.125f;
      }
    }
    __syncthreads();

    // online softmax row stats (one thread per q row)
    if (tid < 64) {
      int row = tid;
      float mold = rowM[row];
      float mmax = mold;
      for (int j = 0; j < 64; ++j) {
        float v = (kv0 + j < len) ? Ssm[row][j] : -1e30f;
        mmax = fmaxf(mmax, v);
      }
      float alpha = __expf(mold - mmax);
      float sum = 0.f;
      for (int j = 0; j < 64; ++j) {
        float v = (kv0 + j < len) ? Ssm[row][j] : -1e30f;
        sum += __expf(v - mmax);
      }
      rowL[row] = rowL[row] * alpha + sum;
      rowM[row] = mmax;
      rowA[row] = alpha;
    }
    __syncthreads();

    // write P tile (bf16) and rescale O accumulators
    {
      int row = tid >> 2, c0 = (tid & 3) * 16;
      float mrow = rowM[row];
#pragma unroll
      for (int j = 0; j < 16; ++j) {
        int c = c0 + j;
        float v = (kv0 + c < len) ? Ssm[row][c] : -1e30f;
        Ps[row][c] = f2bf(__expf(v - mrow));
      }
    }
    {
      int rb = wm * 16 + hi * 8;
#pragma unroll
      for (int r = 0; r < 8; ++r) {
        float a = rowA[rb + r];
        o0[r] *= a; o1[r] *= a;
      }
    }
    __syncthreads();

    // O += P @ V
#pragma unroll
    for (int kk = 0; kk < 64; kk += 32) {
      ABfrag af, b0, b1;
      int am = wm * 16 + ml;
      int ak = kk + hi * 8;
#pragma unroll
      for (int j = 0; j < 4; ++j) {
        af.u[j]     = *(const uint32*)&Ps[am][ak + 2 * j];
        af.u[4 + j] = *(const uint32*)&Ps[am][ak + 16 + 2 * j];
      }
      int bk = kk + hi * 8;
      int bn0 = wn * 32 + ml, bn1 = bn0 + 16;
#pragma unroll
      for (int j = 0; j < 8; ++j) {
        b0.u[j] = *(const uint32*)&Vs[bn0][bk + 2 * j + (j >= 4 ? 8 : 0)];
        b1.u[j] = *(const uint32*)&Vs[bn1][bk + 2 * j + (j >= 4 ? 8 : 0)];
      }
      o0 = __builtin_amdgcn_wmma_f32_16x16x32_bf16(false, af.v, false, b0.v, (short)0, o0, false, false);
      o1 = __builtin_amdgcn_wmma_f32_16x16x32_bf16(false, af.v, false, b1.v, (short)0, o1, false, false);
    }
    __syncthreads();
  }

  // normalize by row sums and write bf16 attention output
  {
    int rb = wm * 16 + hi * 8;
    int c0 = wn * 32 + ml;
#pragma unroll
    for (int r = 0; r < 8; ++r) {
      float linv = 1.0f / rowL[rb + r];
      int orow = (baseRow + q0 + rb + r) * D_ + dcol0;
      O[orow + c0]      = f2bf(o0[r] * linv);
      O[orow + c0 + 16] = f2bf(o1[r] * linv);
    }
  }
}

// ---------------------------------------------------------------------------
// LayerNorm over D with fused residual add: y = LN(acc + res) * g + b.
// Writes fp32 (residual / final output) and bf16 (next GEMM operand).
// ---------------------------------------------------------------------------
__global__ __launch_bounds__(256) void ln_kernel(
    const float* __restrict__ acc, const float* __restrict__ res,
    const float* __restrict__ g, const float* __restrict__ bta,
    float* __restrict__ xout, u16* __restrict__ xbf) {
  __shared__ float red[256], red2[256];
  __shared__ float s_mu, s_rstd;
  int row = blockIdx.x, tid = threadIdx.x;
  int base = row * D_;
  int c = tid * 4;
  float4 a = *(const float4*)(acc + base + c);
  float4 r = *(const float4*)(res + base + c);
  float v0 = a.x + r.x, v1 = a.y + r.y, v2 = a.z + r.z, v3 = a.w + r.w;
  red[tid]  = v0 + v1 + v2 + v3;
  red2[tid] = v0 * v0 + v1 * v1 + v2 * v2 + v3 * v3;
  __syncthreads();
  for (int off = 128; off > 0; off >>= 1) {
    if (tid < off) { red[tid] += red[tid + off]; red2[tid] += red2[tid + off]; }
    __syncthreads();
  }
  if (tid == 0) {
    float mu = red[0] * (1.0f / D_);
    float var = red2[0] * (1.0f / D_) - mu * mu;
    s_mu = mu;
    s_rstd = rsqrtf(var + 1e-5f);
  }
  __syncthreads();
  float mu = s_mu, rstd = s_rstd;
  float4 gg = *(const float4*)(g + c);
  float4 bb = *(const float4*)(bta + c);
  float y0 = (v0 - mu) * rstd * gg.x + bb.x;
  float y1 = (v1 - mu) * rstd * gg.y + bb.y;
  float y2 = (v2 - mu) * rstd * gg.z + bb.z;
  float y3 = (v3 - mu) * rstd * gg.w + bb.w;
  *(float4*)(xout + base + c) = make_float4(y0, y1, y2, y3);
  uint32* xo = (uint32*)(xbf + base + c);
  xo[0] = (uint32)f2bf(y0) | ((uint32)f2bf(y1) << 16);
  xo[1] = (uint32)f2bf(y2) | ((uint32)f2bf(y3) << 16);
}

// ---------------------------------------------------------------------------
extern "C" void kernel_launch(void* const* d_in, const int* in_sizes, int n_in,
                              void* d_out, int out_size, void* d_ws, size_t ws_size,
                              hipStream_t stream) {
  (void)in_sizes; (void)n_in; (void)out_size; (void)ws_size;
  const float* x_in  = (const float*)d_in[0];
  const int* lengths = (const int*)d_in[1];
  const float* emb_w = (const float*)d_in[2];
  const float* emb_b = (const float*)d_in[3];
  const float* qw = (const float*)d_in[4];
  const float* qb = (const float*)d_in[5];
  const float* kw = (const float*)d_in[6];
  const float* kb = (const float*)d_in[7];
  const float* vw = (const float*)d_in[8];
  const float* vb = (const float*)d_in[9];
  const float* ow = (const float*)d_in[10];
  const float* ob = (const float*)d_in[11];
  const float* f1w = (const float*)d_in[12];
  const float* f1b = (const float*)d_in[13];
  const float* f2w = (const float*)d_in[14];
  const float* f2b = (const float*)d_in[15];
  const float* ln_g = (const float*)d_in[16];
  const float* ln_b = (const float*)d_in[17];

  const int M = B_ * S_;  // 8192
  char* ws = (char*)d_ws;
  size_t off = 0;
  auto alloc = [&](size_t bytes) {
    void* p = ws + off;
    off += (bytes + 255) & ~(size_t)255;
    return p;
  };
  float* xres = (float*)alloc((size_t)M * D_ * 4);
  float* obuf = (float*)alloc((size_t)M * D_ * 4);
  u16* xbf = (u16*)alloc((size_t)M * D_ * 2);
  u16* qbf = (u16*)alloc((size_t)M * D_ * 2);
  u16* kbf = (u16*)alloc((size_t)M * D_ * 2);
  u16* vbf = (u16*)alloc((size_t)M * D_ * 2);
  u16* abf = (u16*)alloc((size_t)M * D_ * 2);
  u16* hbf = (u16*)alloc((size_t)M * FF_ * 2);

  dim3 blk(256);
  dim3 gD(D_ / 128, M / 128);
  dim3 gF(FF_ / 128, M / 128);

  // embedding: x @ emb_w + emb_b  (fp32 A)
  gemm_kernel<false><<<gD, blk, 0, stream>>>(x_in, emb_w, emb_b, xres, nullptr,
                                             M, D_, P_, 0);
  // RoPE in place, produce bf16 operand
  rope_kernel<<<M, blk, 0, stream>>>(xres, xbf, lengths);

  for (int i = 0; i < L_; ++i) {
    const float* qwi = qw + (size_t)i * D_ * D_;
    const float* kwi = kw + (size_t)i * D_ * D_;
    const float* vwi = vw + (size_t)i * D_ * D_;
    const float* owi = ow + (size_t)i * D_ * D_;
    gemm_kernel<true><<<gD, blk, 0, stream>>>(xbf, qwi, qb + i * D_, nullptr, qbf,
                                              M, D_, D_, 0);
    gemm_kernel<true><<<gD, blk, 0, stream>>>(xbf, kwi, kb + i * D_, nullptr, kbf,
                                              M, D_, D_, 0);
    gemm_kernel<true><<<gD, blk, 0, stream>>>(xbf, vwi, vb + i * D_, nullptr, vbf,
                                              M, D_, D_, 0);
    flash_kernel<<<dim3(S_ / 64, H_, B_), blk, 0, stream>>>(qbf, kbf, vbf, abf,
                                                            lengths);
    gemm_kernel<true><<<gD, blk, 0, stream>>>(abf, owi, ob + i * D_, obuf, nullptr,
                                              M, D_, D_, 0);
    ln_kernel<<<M, blk, 0, stream>>>(obuf, xres, ln_g + i * D_, ln_b + i * D_,
                                     xres, xbf);
    gemm_kernel<true><<<gF, blk, 0, stream>>>(xbf, f1w + (size_t)i * D_ * FF_,
                                              f1b + i * FF_, nullptr, hbf,
                                              M, FF_, D_, /*ReLU*/ 1);
    gemm_kernel<true><<<gD, blk, 0, stream>>>(hbf, f2w + (size_t)i * FF_ * D_,
                                              f2b + i * D_, obuf, nullptr,
                                              M, D_, FF_, 0);
    float* lnout = (i == L_ - 1) ? (float*)d_out : xres;
    ln_kernel<<<M, blk, 0, stream>>>(obuf, xres, ln_g + i * D_, ln_b + i * D_,
                                     lnout, xbf);
  }
}
